// ContrastiveSparsityLoss_91225105367306
// MI455X (gfx1250) — matrix operand order repaired and checked by reading, not compiled, not run on hardware
//
#include <hip/hip_runtime.h>
#include <math.h>

typedef __attribute__((ext_vector_type(16))) _Float16 v16h;
typedef __attribute__((ext_vector_type(8)))  _Float16 v8h;
typedef __attribute__((ext_vector_type(8)))  float    v8f;

#define EPS_BN   1e-5f
#define NEGV     (-1e9f)
#define TEMP_INV 10.0f   // 1/0.1

// ---------- helpers ----------

__device__ __forceinline__ float red16(float v) {
  // reduce across the 16 lanes of a lane-half (xor masks keep bit4 fixed)
  v += __shfl_xor(v, 1);
  v += __shfl_xor(v, 2);
  v += __shfl_xor(v, 4);
  v += __shfl_xor(v, 8);
  return v;
}

__device__ __forceinline__ v16h join16(v8h lo, v8h hi) {
  return __builtin_shufflevector(lo, hi, 0, 1, 2, 3, 4, 5, 6, 7,
                                         8, 9, 10, 11, 12, 13, 14, 15);
}

// ---------- tiny utility kernels ----------

__global__ void zero_f32(float* __restrict__ p, int n) {
  int i = blockIdx.x * blockDim.x + threadIdx.x;
  if (i < n) p[i] = 0.f;
}

// Pack conv3x3 weights [64][CIN][3][3] f32 -> A1 [64][CIN*16] f16,
// tap index = dy*4+dx (dx==3 / dy==3 slots are zero padding) so that one
// 16-wide K group == one input channel's taps.
__global__ void pack_w1(const float* __restrict__ w1, _Float16* __restrict__ A1, int CIN) {
  int t = blockIdx.x * blockDim.x + threadIdx.x;
  int K16 = CIN * 16;
  int total = 64 * K16;
  if (t >= total) return;
  int co  = t / K16;
  int rem = t - co * K16;
  int ci  = rem >> 4;
  int s   = rem & 15;
  int dy = s >> 2, dx = s & 3;
  float v = (dy < 3 && dx < 3) ? w1[((co * CIN + ci) * 3 + dy) * 3 + dx] : 0.f;
  A1[t] = (_Float16)v;
}

__global__ void pack_w2(const float* __restrict__ w2, _Float16* __restrict__ A2) {
  int t = blockIdx.x * blockDim.x + threadIdx.x;
  if (t < 128 * 64) A2[t] = (_Float16)w2[t];
}

// ---------- conv3x3 as implicit GEMM with WMMA ----------
// Block: 128 threads (4 waves). Tile: 1 output row x 64 pixels, all 64 Co.
// Wave wv computes Co[0..63] x pixels[wv*16 .. wv*16+15].
// K dim = CIN*16 (padded taps); one 16x16x32 WMMA chunk = 2 input channels.
template <int CIN>
__global__ __launch_bounds__(128) void conv3x3_wmma(
    const float* __restrict__ x,        // [5][CIN][256][256]
    const _Float16* __restrict__ A1,    // [64][CIN*16]
    _Float16* __restrict__ y,           // [5][64][256][256] f16
    float* __restrict__ stats1) {       // [2][5][64]  sum / sumsq
  constexpr int K16 = CIN * 16;
  constexpr int XS  = CIN * 3 * 66;     // halo tile, f32
  __shared__ float xs[XS];
  __shared__ float sstat[128];

  const int tid   = threadIdx.x;
  const int lane  = tid & 31;
  const int wv    = tid >> 5;
  const int wbase = blockIdx.x * 64;
  const int h     = blockIdx.y;
  const int n     = blockIdx.z;

  // Stage padded input tile into LDS (f32; cvt to f16 at fragment build time)
  for (int i = tid; i < XS; i += 128) {
    int ci  = i / 198;
    int rem = i - ci * 198;
    int dy  = rem / 66;
    int col = rem - dy * 66;
    int gy = h - 1 + dy;
    int gx = wbase - 1 + col;
    float v = 0.f;
    if (gy >= 0 && gy < 256 && gx >= 0 && gx < 256)
      v = x[(((size_t)n * CIN + ci) * 256 + gy) * 256 + gx];
    xs[i] = v;
  }
  if (tid < 128) sstat[tid] = 0.f;
  __syncthreads();

  const int pxl    = wv * 16 + (lane & 15);     // local pixel 0..63
  const int hiHalf = (lane & 16) ? 1 : 0;

  v8f acc[4] = {};
  for (int kc = 0; kc < CIN / 2; ++kc) {
    // B fragment: 16 consecutive K per lane = one channel's padded taps
    const int ci = 2 * kc + hiHalf;
    const float* xr = &xs[ci * 198];
    v16h b = {};
#pragma unroll
    for (int dy = 0; dy < 3; ++dy) {
      b[dy * 4 + 0] = (_Float16)xr[dy * 66 + pxl + 0];
      b[dy * 4 + 1] = (_Float16)xr[dy * 66 + pxl + 1];
      b[dy * 4 + 2] = (_Float16)xr[dy * 66 + pxl + 2];
    }
    // A fragments from L2-resident packed weights
    const int co0 = lane & 15;
    const int kb  = kc * 32 + hiHalf * 16;
#pragma unroll
    for (int mb = 0; mb < 4; ++mb) {
      const _Float16* ar = A1 + (size_t)(mb * 16 + co0) * K16 + kb;
      v16h a = join16(*(const v8h*)ar, *(const v8h*)(ar + 8));
      acc[mb] = __builtin_amdgcn_wmma_f32_16x16x32_f16(
          false, a, false, b, (short)0, acc[mb], false, false);
    }
  }

  // Epilogue: store f16 feature map + accumulate BN1 statistics
  const int colOut = wbase + pxl;
#pragma unroll
  for (int mb = 0; mb < 4; ++mb) {
#pragma unroll
    for (int r = 0; r < 8; ++r) {
      float v = acc[mb][r];
      int co = mb * 16 + r + hiHalf * 8;
      y[(((size_t)n * 64 + co) * 256 + h) * 256 + colOut] = (_Float16)v;
      float s  = red16(v);
      float s2 = red16(v * v);
      if ((lane & 15) == 0) {
        atomicAdd(&sstat[co], s);
        atomicAdd(&sstat[64 + co], s2);
      }
    }
  }
  __syncthreads();
  if (tid < 128)
    atomicAdd(&stats1[(tid >> 6) * 320 + n * 64 + (tid & 63)], sstat[tid]);
}

// ---------- BN stats -> scale/bias ----------
// stats [2][5][C]; bn out [2][5][C] (scale, bias). batch!=0 -> axes (0,2,3).
__global__ void bn_finalize(const float* __restrict__ stats,
                            const float* __restrict__ gamma,
                            const float* __restrict__ beta,
                            float* __restrict__ bn, int C, int batch) {
  int c = threadIdx.x;
  if (c >= C) return;
  const float inv_px = 1.0f / 65536.0f;
  if (batch) {
    float s = 0.f, s2 = 0.f;
    for (int nn = 0; nn < 5; ++nn) {
      s  += stats[nn * C + c];
      s2 += stats[5 * C + nn * C + c];
    }
    float m   = s * inv_px * 0.2f;
    float var = s2 * inv_px * 0.2f - m * m;
    float sc  = gamma[c] * rsqrtf(var + EPS_BN);
    float bi  = beta[c] - m * sc;
    for (int nn = 0; nn < 5; ++nn) {
      bn[nn * C + c] = sc;
      bn[5 * C + nn * C + c] = bi;
    }
  } else {
    for (int nn = 0; nn < 5; ++nn) {
      float s  = stats[nn * C + c];
      float s2 = stats[5 * C + nn * C + c];
      float m   = s * inv_px;
      float var = s2 * inv_px - m * m;
      float sc  = gamma[c] * rsqrtf(var + EPS_BN);
      bn[nn * C + c] = sc;
      bn[5 * C + nn * C + c] = beta[c] - m * sc;
    }
  }
}

// ---------- 1x1 conv (GEMM 128x64 @ 64xPix) with WMMA ----------
// PASS2=false: accumulate h2 sum/sumsq (BN2 stats).
// PASS2=true : apply BN2+ReLU and accumulate GAP sums.
template <bool PASS2>
__global__ __launch_bounds__(128) void gemm1x1_wmma(
    const _Float16* __restrict__ yh,    // [5][64][65536]
    const _Float16* __restrict__ A2,    // [128][64]
    const float* __restrict__ bn1,      // [2][5][64]
    const float* __restrict__ bn2,      // [2][5][128] (PASS2 only)
    float* __restrict__ outAcc) {       // stats2 [2][5][128] or gap [5][128]
  __shared__ __align__(16) _Float16 a2s[128][64];
  __shared__ __align__(16) _Float16 h1s[64][64];   // [pixel][ci] transposed
  __shared__ float sred[256];

  const int tid  = threadIdx.x;
  const int lane = tid & 31;
  const int wv   = tid >> 5;
  const int n    = blockIdx.y;
  const size_t pixbase = (size_t)blockIdx.x * 64;

  for (int i = tid; i < 128 * 64; i += 128) ((_Float16*)a2s)[i] = A2[i];
  for (int i = tid; i < 64 * 64; i += 128) {
    int ci = i >> 6, p = i & 63;                 // consecutive tid -> consecutive pixel
    float v = (float)yh[((size_t)n * 64 + ci) * 65536 + pixbase + p];
    v = fmaxf(v * bn1[n * 64 + ci] + bn1[320 + n * 64 + ci], 0.f);
    h1s[p][ci] = (_Float16)v;
  }
  for (int i = tid; i < 256; i += 128) sred[i] = 0.f;
  __syncthreads();

  const int pxl    = wv * 16 + (lane & 15);
  const int hiHalf = (lane & 16) ? 1 : 0;

  v8f acc[8] = {};
#pragma unroll
  for (int kc = 0; kc < 2; ++kc) {
    const _Float16* br = &h1s[pxl][kc * 32 + hiHalf * 16];
    v16h b = join16(*(const v8h*)br, *(const v8h*)(br + 8));
    const int kb = kc * 32 + hiHalf * 16;
#pragma unroll
    for (int mb = 0; mb < 8; ++mb) {
      const _Float16* ar = &a2s[mb * 16 + (lane & 15)][kb];
      v16h a = join16(*(const v8h*)ar, *(const v8h*)(ar + 8));
      acc[mb] = __builtin_amdgcn_wmma_f32_16x16x32_f16(
          false, a, false, b, (short)0, acc[mb], false, false);
    }
  }

#pragma unroll
  for (int mb = 0; mb < 8; ++mb) {
#pragma unroll
    for (int r = 0; r < 8; ++r) {
      float v = acc[mb][r];
      int d = mb * 16 + r + hiHalf * 8;
      if (PASS2) {
        float hv = fmaxf(v * bn2[n * 128 + d] + bn2[640 + n * 128 + d], 0.f);
        float s = red16(hv);
        if ((lane & 15) == 0) atomicAdd(&sred[d], s);
      } else {
        float s  = red16(v);
        float s2 = red16(v * v);
        if ((lane & 15) == 0) {
          atomicAdd(&sred[d], s);
          atomicAdd(&sred[128 + d], s2);
        }
      }
    }
  }
  __syncthreads();
  if (PASS2) {
    if (tid < 128) atomicAdd(&outAcc[n * 128 + tid], sred[tid]);
  } else {
    for (int i = tid; i < 256; i += 128)
      atomicAdd(&outAcc[(i >> 7) * 640 + n * 128 + (i & 127)], sred[i]);
  }
}

// ---------- GAP -> L2-normalized embedding ----------
__global__ void vec_norm(const float* __restrict__ gap, float* __restrict__ out) {
  __shared__ float red[128];
  int n = blockIdx.x, d = threadIdx.x;
  float v = gap[n * 128 + d] * (1.0f / 65536.0f);
  red[d] = v * v;
  __syncthreads();
  for (int s = 64; s > 0; s >>= 1) {
    if (d < s) red[d] += red[d + s];
    __syncthreads();
  }
  float nrm = fmaxf(sqrtf(red[0]), 1e-12f);
  out[n * 128 + d] = v / nrm;
}

// ---------- final contrastive loss ----------
__global__ void loss_kernel(const float* __restrict__ Q,   // [3][5][128]
                            const float* __restrict__ Kv,  // [3][5][128]
                            const int* __restrict__ mask,  // [4][128][128]
                            float* __restrict__ out) {
  __shared__ int   scnt[16];
  __shared__ float sloss[16];
  int tid = threadIdx.x;
  if (tid < 16) { scnt[tid] = 0; sloss[tid] = 0.f; }
  __syncthreads();

  int lc[3][5] = {};
  for (int e = tid; e < 4 * 128 * 128; e += 256) {
    int a = e >> 14;
    int m = mask[e];
    if (m >= 1 && m <= 3) lc[m - 1][a + 1]++;
  }
  for (int g = 0; g < 3; ++g)
    for (int i = 1; i < 5; ++i)
      if (lc[g][i]) atomicAdd(&scnt[g * 5 + i], lc[g][i]);
  __syncthreads();

  if (tid < 15) {
    int g = tid / 5, i = tid % 5;
    const float* q = Q + (g * 5 + i) * 128;
    float l[9];
    {
      const float* k = Kv + (g * 5 + i) * 128;
      float s = 0.f;
      for (int d = 0; d < 128; ++d) s += q[d] * k[d];
      l[0] = s;
    }
    for (int j = 0; j < 5; ++j) {
      if (j == i) { l[1 + j] = NEGV; continue; }
      const float* k = Kv + (g * 5 + j) * 128;
      float s = 0.f;
      for (int d = 0; d < 128; ++d) s += q[d] * k[d];
      l[1 + j] = s;
    }
    for (int hh = 0; hh < 3; ++hh) {
      if (hh == g) { l[6 + hh] = NEGV; continue; }
      const float* k = Kv + (hh * 5 + i) * 128;
      float s = 0.f;
      for (int d = 0; d < 128; ++d) s += q[d] * k[d];
      l[6 + hh] = s;
    }
    float mx = -1e30f;
    for (int t = 0; t < 9; ++t) { l[t] *= TEMP_INV; mx = fmaxf(mx, l[t]); }
    float se = 0.f;
    for (int t = 0; t < 9; ++t) se += expf(l[t] - mx);
    sloss[tid] = -l[0] + mx + logf(se);
  }
  __syncthreads();
  if (tid == 0) {
    float tot = 0.f, accv = 0.f;
    for (int t = 0; t < 15; ++t) {
      tot  += (float)scnt[t];
      accv += (float)scnt[t] * sloss[t];
    }
    out[0] = (tot > 0.f) ? accv / fmaxf(tot, 1.0f) : 0.f;
  }
}

// ---------- orchestration ----------

extern "C" void kernel_launch(void* const* d_in, const int* in_sizes, int n_in,
                              void* d_out, int out_size, void* d_ws, size_t ws_size,
                              hipStream_t stream) {
  (void)in_sizes; (void)n_in; (void)out_size; (void)ws_size;
  const int CINg[3] = {64, 32, 16};
  char* ws = (char*)d_ws;

  // workspace layout (bytes)
  size_t off = 0;
  size_t offA1[3], offA2[3];
  for (int g = 0; g < 3; ++g) { offA1[g] = off; off += (size_t)64 * CINg[g] * 16 * 2; }
  for (int g = 0; g < 3; ++g) { offA2[g] = off; off += (size_t)128 * 64 * 2; }
  size_t offStats1 = off; off += 2 * 5 * 64 * 4;    // [2][5][64]
  size_t offStats2 = off; off += 2 * 5 * 128 * 4;   // [2][5][128]
  size_t offGap    = off; off += 5 * 128 * 4;       // [5][128]
  size_t offBn1    = off; off += 2 * 5 * 64 * 4;
  size_t offBn2    = off; off += 2 * 5 * 128 * 4;
  size_t offQ      = off; off += 3 * 5 * 128 * 4;
  size_t offK      = off; off += 3 * 5 * 128 * 4;
  off = (off + 511) & ~(size_t)511;
  size_t offY = off;                                // f16 [5][64][256][256]

  float* stats1 = (float*)(ws + offStats1);
  float* stats2 = (float*)(ws + offStats2);
  float* gap    = (float*)(ws + offGap);
  float* bn1    = (float*)(ws + offBn1);
  float* bn2    = (float*)(ws + offBn2);
  float* Qb     = (float*)(ws + offQ);
  float* Kb     = (float*)(ws + offK);
  _Float16* yh  = (_Float16*)(ws + offY);

  // pack weights to f16 WMMA-friendly layouts
  for (int g = 0; g < 3; ++g) {
    const float* w1 = (const float*)d_in[g * 8 + 2];
    const float* w2 = (const float*)d_in[g * 8 + 5];
    int tot = 64 * CINg[g] * 16;
    pack_w1<<<(tot + 255) / 256, 256, 0, stream>>>(w1, (_Float16*)(ws + offA1[g]), CINg[g]);
    pack_w2<<<(128 * 64 + 255) / 256, 256, 0, stream>>>(w2, (_Float16*)(ws + offA2[g]));
  }

  for (int g = 0; g < 3; ++g) {
    const float* g1g = (const float*)d_in[g * 8 + 3];
    const float* b1g = (const float*)d_in[g * 8 + 4];
    const float* g2g = (const float*)d_in[g * 8 + 6];
    const float* b2g = (const float*)d_in[g * 8 + 7];
    const _Float16* A1 = (const _Float16*)(ws + offA1[g]);
    const _Float16* A2 = (const _Float16*)(ws + offA2[g]);
    for (int path = 0; path < 2; ++path) {
      const float* xin = (const float*)d_in[g * 8 + path];  // 0: sparse, 1: dense
      int batch = path;  // dense key path uses batch BN stats
      // zero stats1 + stats2 + gap (contiguous: 640+1280+640 = 2560 floats)
      zero_f32<<<(2560 + 255) / 256, 256, 0, stream>>>(stats1, 2560);
      dim3 cgrid(4, 256, 5);
      if (g == 0)      conv3x3_wmma<64><<<cgrid, 128, 0, stream>>>(xin, A1, yh, stats1);
      else if (g == 1) conv3x3_wmma<32><<<cgrid, 128, 0, stream>>>(xin, A1, yh, stats1);
      else             conv3x3_wmma<16><<<cgrid, 128, 0, stream>>>(xin, A1, yh, stats1);
      bn_finalize<<<1, 64, 0, stream>>>(stats1, g1g, b1g, bn1, 64, batch);
      dim3 ggrid(1024, 5);
      gemm1x1_wmma<false><<<ggrid, 128, 0, stream>>>(yh, A2, bn1, bn2, stats2);
      bn_finalize<<<1, 128, 0, stream>>>(stats2, g2g, b2g, bn2, 128, batch);
      gemm1x1_wmma<true><<<ggrid, 128, 0, stream>>>(yh, A2, bn1, bn2, gap);
      float* vout = (path == 0 ? Qb : Kb) + g * 5 * 128;
      vec_norm<<<5, 128, 0, stream>>>(gap, vout);
    }
  }

  loss_kernel<<<1, 256, 0, stream>>>(Qb, Kb, (const int*)d_in[24], (float*)d_out);
}
